// KrigingObsAdapter_18966575579407
// MI455X (gfx1250) — matrix operand order repaired and checked by reading, not compiled, not run on hardware
//
#include <hip/hip_runtime.h>
#include <hip/hip_bf16.h>
#include <math.h>

// ---------------------------------------------------------------------------
// Problem constants (from reference)
// ---------------------------------------------------------------------------
#define BB   256
#define TT   365
#define DD   256
#define SS   32
#define FF   8
#define KK   64
#define NN   1024
#define BT   (BB*TT)       // 93440  (= 730 * 128)
#define BN   (BB*NN)       // 262144 (= 2048 * 128)
#define EPSW 1e-8f

// d_in layout (setup_inputs() insertion order, params dict insertion order,
// each linear = (w, b)):
//  0 hidden (B,T,D)   1 static (B,S)   2 time (B,T,F)   3 bank (N,34)
//  4 Wpg0(32,128) 5 bpg0  6 Wpg1(128,64) 7 bpg1
//  8 Wvp(32,64)   9 bvp
// 10 Wrg0(128,64) 11 brg0  12 wrg1(64,1) 13 brg1
// 14 Wlt(8,256)   15 blt
// 16 Wls(32,256)  17 bls
// 18 Wlc0(768,512) 19 blc0  20 Wlc1(512,256) 21 blc1
// 22 ln_gamma(256) 23 ln_beta(256)
// 24 Wk0(34,64) 25 bk0  26 Wk1(64,64) 27 bk1  28 wk2(64,1) 29 bk2
// 30 Wpo0(35,128) 31 bpo0  32 Wpo1(128,128) 33 bpo1  34 Wpo2(128,128) 35 bpo2
// 36 Wro0(128,128) 37 bro0  38 Wro1(128,256) 39 bro1
// 40 Wg0(2,32) 41 bg0  42 Wg1(32,32) 43 bg1  44 wg2(32,1) 45 bg2

typedef __attribute__((ext_vector_type(16))) _Float16 v16h;
typedef __attribute__((ext_vector_type(8)))  float    v8f;
typedef __attribute__((ext_vector_type(4)))  unsigned int v4u;
typedef __attribute__((ext_vector_type(4)))  unsigned int u32x4;
typedef __attribute__((ext_vector_type(8)))  int          i32x8;
typedef __attribute__((ext_vector_type(4)))  int          i32x4;

union FragH { v16h h; v4u u[2]; };

__device__ __forceinline__ float gelu_f(float x) {
    return 0.5f * x * (1.0f + erff(x * 0.70710678118654752f));
}

// ---------------------------------------------------------------------------
// Tensor Data Mover staging (gfx1250): load a 2-D f16 tile (tile_h rows of
// tile_w elements, row stride = row_stride elements) from global memory into
// LDS at byte offset lds_byte. Tracked by TENSORcnt. Guarded so the host pass
// and builtin-less toolchains fall back to direct global loads.
// ---------------------------------------------------------------------------
#if defined(__HIP_DEVICE_COMPILE__) && \
    __has_builtin(__builtin_amdgcn_tensor_load_to_lds) && \
    __has_builtin(__builtin_amdgcn_s_wait_tensorcnt)
#define USE_TDM 1
#endif

__device__ __forceinline__ void tdm_load_tile_f16(const _Float16* gsrc, unsigned lds_byte,
                                                  unsigned tile_w, unsigned tile_h,
                                                  unsigned row_stride) {
#ifdef USE_TDM
    unsigned long long ga = (unsigned long long)(const void*)gsrc;
    // D# group 0: count=1 | lds_addr | global_addr[56:0] | type=2
    u32x4 g0;
    g0.x = 1u;
    g0.y = lds_byte;
    g0.z = (unsigned)(ga & 0xFFFFFFFFu);
    g0.w = (unsigned)((ga >> 32) & 0x01FFFFFFu) | (2u << 30);
    // D# group 1: data_size=2B; tensor_dim0=row_stride; tensor_dim1=tile_h;
    //             tile_dim0=tile_w; tile_dim1=tile_h; tensor_dim0_stride=row_stride
    i32x8 g1;
    g1[0] = (int)(1u << 16);                                  // data_size code 1 (2 bytes)
    g1[1] = (int)((row_stride & 0xFFFFu) << 16);              // tensor_dim0[15:0] @ bits63:48
    g1[2] = (int)(((row_stride >> 16) & 0xFFFFu) |            // tensor_dim0[31:16]
                  ((tile_h & 0xFFFFu) << 16));                // tensor_dim1[15:0]
    g1[3] = (int)(((tile_h >> 16) & 0xFFFFu) |                // tensor_dim1[31:16]
                  ((tile_w & 0xFFFFu) << 16));                // tile_dim0
    g1[4] = (int)(tile_h & 0xFFFFu);                          // tile_dim1 (tile_dim2=0)
    g1[5] = (int)row_stride;                                  // tensor_dim0_stride[31:0]
    g1[6] = 0;                                                // stride hi / dim1_stride
    g1[7] = 0;
    i32x4 gz = {0, 0, 0, 0};
#if __clang_major__ >= 23
    i32x8 gz8 = {0, 0, 0, 0, 0, 0, 0, 0};
    __builtin_amdgcn_tensor_load_to_lds(g0, g1, gz, gz, gz8, 0);
#else
    __builtin_amdgcn_tensor_load_to_lds(g0, g1, gz, gz, 0);
#endif
#else
    (void)gsrc; (void)lds_byte; (void)tile_w; (void)tile_h; (void)row_stride;
#endif
}

__device__ __forceinline__ void tdm_wait() {
#ifdef USE_TDM
    __builtin_amdgcn_s_wait_tensorcnt(0);
#endif
}

// ---------------------------------------------------------------------------
// Generic WMMA f16 GEMM:  C = act(A(MxK,f16) @ W(KxN,f16) + bias)
//   - block: 256 threads = 8 waves; each wave owns a 16-row strip; block = 128 rows
//   - A tile (128 x 32) staged into LDS by the Tensor Data Mover (wave 0),
//     fragments read as 2x ds_load_b128 per the CDNA5 16-bit A 16x32 layout
//   - W k-tile (32 x BLKN) staged transposed in LDS -> B frags are 2x ds_load_b128
//   - TRB: W given as (N x K) row-major (used for q @ k^T)
// Requirements: M % 128 == 0, K % 32 == 0, lda/ldw/ldc element strides.
// ---------------------------------------------------------------------------
template<int BLKN, int ACT, bool TRB, typename OutT>
__global__ __launch_bounds__(256)
void gemm_f16(const _Float16* __restrict__ A, int lda,
              const _Float16* __restrict__ W, int ldw,
              const float* __restrict__ bias,
              OutT* __restrict__ C, int ldc, int K)
{
    __shared__ _Float16 atile[128 * 32];  // TDM-staged A tile (row-major, 8 KB)
    __shared__ _Float16 wt[BLKN * 40];    // [n][k], k-stride padded to 40 halves

    const int tid  = threadIdx.x;
    const int wave = tid >> 5;
    const int lane = tid & 31;
    const int m    = lane & 15;
    const int half = (lane < 16) ? 0 : 1;

    const int rowB = blockIdx.x * 128;
    const int row0 = rowB + wave * 16;
    const int col0 = blockIdx.y * BLKN;
    const size_t arow = (size_t)(row0 + m);

    // low 32 bits of the flat LDS pointer = LDS byte offset (ISA 10.2)
    const unsigned lds_atile = (unsigned)(unsigned long long)(const void*)atile;

    v8f acc[BLKN / 16];
#pragma unroll
    for (int i = 0; i < BLKN / 16; ++i)
#pragma unroll
        for (int r = 0; r < 8; ++r) acc[i][r] = 0.0f;

    for (int k0 = 0; k0 < K; k0 += 32) {
        __syncthreads();   // previous iteration's LDS reads complete
        // async A-tile DMA: global (128 x 32 f16 tile of A) -> LDS, one issue per block
        if (wave == 0)
            tdm_load_tile_f16(A + (size_t)rowB * lda + k0, lds_atile, 32u, 128u, (unsigned)lda);
        // cooperative transpose-load of the 32 x BLKN weight tile into LDS
        for (int idx = tid; idx < 32 * BLKN; idx += 256) {
            int kk, nn;
            if (TRB) { kk = idx & 31;  nn = idx >> 5; }     // coalesce along k
            else     { nn = idx % BLKN; kk = idx / BLKN; }  // coalesce along n
            _Float16 wv = TRB ? W[(size_t)(col0 + nn) * ldw + (k0 + kk)]
                              : W[(size_t)(k0 + kk) * ldw + (col0 + nn)];
            wt[nn * 40 + kk] = wv;
        }
        // prefetch next k-step tiles while this one is consumed
        if (k0 + 32 < K) {
            __builtin_prefetch(A + arow * lda + k0 + 32, 0, 1);
            __builtin_prefetch(TRB ? (W + (size_t)(col0 + (tid % BLKN)) * ldw + k0 + 32)
                                   : (W + (size_t)(k0 + 32 + (tid & 31)) * ldw + col0),
                               0, 1);
        }
        if (wave == 0) tdm_wait();
        __syncthreads();   // W stores + A TDM visible

        // A fragment: lanes 0-15 -> k [0..7]+[16..23], lanes 16-31 -> k [8..15]+[24..31]
        FragH a;
#ifdef USE_TDM
        const _Float16* ap = &atile[(wave * 16 + m) * 32 + half * 8];
#else
        const _Float16* ap = A + arow * lda + k0 + half * 8;
#endif
        a.u[0] = *(const v4u*)(ap);
        a.u[1] = *(const v4u*)(ap + 16);

#pragma unroll
        for (int nt = 0; nt < BLKN / 16; ++nt) {
            // B fragment: col = lane%16, k [0..7]+[16..23] (lanes 0-15) / [8..15]+[24..31] (16-31)
            FragH b;
            const _Float16* wp = &wt[(nt * 16 + m) * 40 + half * 8];
            b.u[0] = *(const v4u*)(wp);
            b.u[1] = *(const v4u*)(wp + 16);
            acc[nt] = __builtin_amdgcn_wmma_f32_16x16x32_f16(
                false, a.h, false, b.h, (short)0, acc[nt], false, false);
        }
    }

    // C/D layout: lane col = lane%16; VGPR r -> row r (lanes 0-15) / r+8 (16-31)
    const int srow = row0 + half * 8;
#pragma unroll
    for (int nt = 0; nt < BLKN / 16; ++nt) {
        const int gcol = col0 + nt * 16 + m;
        const float bv = bias ? bias[gcol] : 0.0f;
#pragma unroll
        for (int r = 0; r < 8; ++r) {
            float v = acc[nt][r] + bv;
            if (ACT == 1) v = gelu_f(v);
            C[(size_t)(srow + r) * ldc + gcol] = (OutT)v;
        }
    }
}

// ---------------------------------------------------------------------------
// Small helper kernels (bandwidth-trivial; plain f32 VALU)
// ---------------------------------------------------------------------------

// f32 -> f16 convert with zero K-padding (dst is Kpad x N row-major)
__global__ void k_convert_pad(const float* __restrict__ src, _Float16* __restrict__ dst,
                              int Ksrc, int N, int Kpad) {
    int i = blockIdx.x * 256 + threadIdx.x;
    if (i >= Kpad * N) return;
    int k = i / N, n = i % N;
    dst[i] = (k < Ksrc) ? (_Float16)src[k * N + n] : (_Float16)0.0f;
}

// c_bank = bank[:, :32] -> f16 (lda 32)
__global__ void k_cbank(const float* __restrict__ bank, _Float16* __restrict__ dst) {
    int i = blockIdx.x * 256 + threadIdx.x;
    if (i >= NN * SS) return;
    int n = i >> 5, s = i & 31;
    dst[i] = (_Float16)bank[n * 34 + s];
}

// Build pairwise token matrices (padded to K=64):
//  Xphi = [diff(32), dist, y, m, 0...]   Xkap = [diff(32), dist, m, 0...]
__global__ __launch_bounds__(256)
void k_tokens(const float* __restrict__ sf, const float* __restrict__ bank,
              _Float16* __restrict__ Xphi, _Float16* __restrict__ Xkap) {
    int row = blockIdx.x * 256 + threadIdx.x;      // BN rows
    int b = row >> 10, n = row & 1023;
    const float* sb = sf + b * SS;
    const float* bk = bank + n * 34;
    size_t off = (size_t)row * 64;
    float d2 = 0.0f;
    for (int s = 0; s < 32; ++s) {
        float diff = bk[s] - sb[s];
        d2 += diff * diff;
        _Float16 h = (_Float16)diff;
        Xphi[off + s] = h;
        Xkap[off + s] = h;
    }
    float dist = sqrtf(d2);
    Xphi[off + 32] = (_Float16)dist;
    Xphi[off + 33] = (_Float16)bk[32];   // y
    Xphi[off + 34] = (_Float16)bk[33];   // m
    Xkap[off + 32] = (_Float16)dist;
    Xkap[off + 33] = (_Float16)bk[33];   // m
    for (int s = 35; s < 64; ++s) Xphi[off + s] = (_Float16)0.0f;
    for (int s = 34; s < 64; ++s) Xkap[off + s] = (_Float16)0.0f;
}

// kappa head: w_raw = softplus(H2 . wk2 + b) * m
__global__ void k_kappa3(const _Float16* __restrict__ H2, const float* __restrict__ w2,
                         const float* __restrict__ b2, const float* __restrict__ bank,
                         float* __restrict__ w_raw) {
    int row = blockIdx.x * 256 + threadIdx.x;
    if (row >= BN) return;
    int n = row & 1023;
    float s = b2[0];
    const _Float16* h = H2 + (size_t)row * 64;
    for (int j = 0; j < 64; ++j) s += (float)h[j] * w2[j];
    float sp = (s > 20.0f) ? s : log1pf(expf(s));
    w_raw[row] = sp * bank[n * 34 + 33];
}

__global__ __launch_bounds__(256)
void k_wsum(const float* __restrict__ w_raw, float* __restrict__ Wsum) {
    __shared__ float red[256];
    int b = blockIdx.x;
    float s = 0.0f;
    for (int i = threadIdx.x; i < NN; i += 256) s += w_raw[b * NN + i];
    red[threadIdx.x] = s; __syncthreads();
    for (int st = 128; st > 0; st >>= 1) {
        if (threadIdx.x < st) red[threadIdx.x] += red[threadIdx.x + st];
        __syncthreads();
    }
    if (threadIdx.x == 0) Wsum[b] = red[0];
}

__global__ __launch_bounds__(256)
void k_meanm(const float* __restrict__ bank, float* __restrict__ vfrac) {
    __shared__ float red[256];
    float s = 0.0f;
    for (int i = threadIdx.x; i < NN; i += 256) s += bank[i * 34 + 33];
    red[threadIdx.x] = s; __syncthreads();
    for (int st = 128; st > 0; st >>= 1) {
        if (threadIdx.x < st) red[threadIdx.x] += red[threadIdx.x + st];
        __syncthreads();
    }
    if (threadIdx.x == 0) vfrac[0] = red[0] / (float)NN;
}

// softmax over 1024 cols with 1/sqrt(K)=0.125 scale; f16 out
__global__ __launch_bounds__(256)
void k_softmax(const float* __restrict__ scores, _Float16* __restrict__ attn) {
    __shared__ float red[256];
    int b = blockIdx.x, t = threadIdx.x;
    float v[4], mx = -1e30f;
    for (int i = 0; i < 4; ++i) {
        v[i] = scores[(size_t)b * NN + t + i * 256] * 0.125f;
        mx = fmaxf(mx, v[i]);
    }
    red[t] = mx; __syncthreads();
    for (int st = 128; st > 0; st >>= 1) {
        if (t < st) red[t] = fmaxf(red[t], red[t + st]);
        __syncthreads();
    }
    mx = red[0]; __syncthreads();
    float s = 0.0f;
    for (int i = 0; i < 4; ++i) { v[i] = expf(v[i] - mx); s += v[i]; }
    red[t] = s; __syncthreads();
    for (int st = 128; st > 0; st >>= 1) {
        if (t < st) red[t] += red[t + st];
        __syncthreads();
    }
    float inv = 1.0f / red[0];
    for (int i = 0; i < 4; ++i)
        attn[(size_t)b * NN + t + i * 256] = (_Float16)(v[i] * inv);
}

// alpha = sigmoid(h1 . wrg1 + b)
__global__ void k_alpha(const _Float16* __restrict__ h1, const float* __restrict__ w,
                        const float* __restrict__ bias, float* __restrict__ alpha) {
    int b = blockIdx.x * 256 + threadIdx.x;
    if (b >= BB) return;
    float s = bias[0];
    for (int j = 0; j < 64; ++j) s += (float)h1[b * 64 + j] * w[j];
    alpha[b] = 1.0f / (1.0f + expf(-s));
}

// gate_net: [log(Wsum+eps), vfrac] -> 32 -> 32 -> sigmoid
__global__ void k_gate(const float* __restrict__ Wsum, const float* __restrict__ vfrac,
                       const float* __restrict__ w0, const float* __restrict__ b0,
                       const float* __restrict__ w1, const float* __restrict__ b1,
                       const float* __restrict__ w2, const float* __restrict__ b2,
                       float* __restrict__ g) {
    int b = blockIdx.x * 256 + threadIdx.x;
    if (b >= BB) return;
    float f0 = logf(Wsum[b] + EPSW), f1 = vfrac[0];
    float h1[32], h2[32];
    for (int j = 0; j < 32; ++j) h1[j] = gelu_f(f0 * w0[j] + f1 * w0[32 + j] + b0[j]);
    for (int j = 0; j < 32; ++j) {
        float s = b1[j];
        for (int i = 0; i < 32; ++i) s += h1[i] * w1[i * 32 + j];
        h2[j] = gelu_f(s);
    }
    float s = b2[0];
    for (int i = 0; i < 32; ++i) s += h2[i] * w2[i];
    g[b] = 1.0f / (1.0f + expf(-s));
}

// agg[b,d] = sum_n w_raw * phi / (Wsum + eps)
__global__ __launch_bounds__(128)
void k_agg(const _Float16* __restrict__ P3, const float* __restrict__ w_raw,
           const float* __restrict__ Wsum, _Float16* __restrict__ agg) {
    int b = blockIdx.x, d = threadIdx.x;
    float s = 0.0f;
    for (int n = 0; n < NN; ++n)
        s += w_raw[b * NN + n] * (float)P3[((size_t)b * NN + n) * 128 + d];
    agg[b * 128 + d] = (_Float16)(s / (Wsum[b] + EPSW));
}

// Xloc = f16 concat [hidden | time @ Wlt + blt | sproj]
__global__ __launch_bounds__(256)
void k_xloc(const float* __restrict__ hidden, const float* __restrict__ timef,
            const float* __restrict__ Wt, const float* __restrict__ bt,
            const float* __restrict__ sproj, _Float16* __restrict__ X) {
    __shared__ float ts[FF];
    int row = blockIdx.x, d = threadIdx.x;
    int b = row / TT;
    if (d < FF) ts[d] = timef[(size_t)row * FF + d];
    __syncthreads();
    float tp = bt[d];
    for (int f = 0; f < FF; ++f) tp += ts[f] * Wt[f * DD + d];
    size_t off = (size_t)row * 768;
    X[off + d]        = (_Float16)hidden[(size_t)row * DD + d];
    X[off + 256 + d]  = (_Float16)tp;
    X[off + 512 + d]  = (_Float16)sproj[b * DD + d];
}

// final: z = loc + alpha*g*dE; layernorm; out = hidden + zn*gamma + beta
__global__ __launch_bounds__(256)
void k_final(const float* __restrict__ hidden, const float* __restrict__ loc,
             const float* __restrict__ dE, const float* __restrict__ alpha,
             const float* __restrict__ g, const float* __restrict__ gamma,
             const float* __restrict__ beta, float* __restrict__ out) {
    __shared__ float red[256];
    int row = blockIdx.x, d = threadIdx.x;
    int b = row / TT;
    size_t idx = (size_t)row * DD + d;
    float z = loc[idx] + alpha[b] * g[b] * dE[b * DD + d];
    red[d] = z; __syncthreads();
    for (int st = 128; st > 0; st >>= 1) {
        if (d < st) red[d] += red[d + st];
        __syncthreads();
    }
    float mu = red[0] / (float)DD; __syncthreads();
    float c = z - mu;
    red[d] = c * c; __syncthreads();
    for (int st = 128; st > 0; st >>= 1) {
        if (d < st) red[d] += red[d + st];
        __syncthreads();
    }
    float var = red[0] / (float)DD;
    float zn = c * rsqrtf(var + 1e-5f);
    out[idx] = hidden[idx] + zn * gamma[d] + beta[d];
}

// ---------------------------------------------------------------------------
// Host launcher
// ---------------------------------------------------------------------------
extern "C" void kernel_launch(void* const* d_in, const int* in_sizes, int n_in,
                              void* d_out, int out_size, void* d_ws, size_t ws_size,
                              hipStream_t stream) {
    (void)in_sizes; (void)n_in; (void)out_size; (void)ws_size;
    const float* I[46];
    for (int i = 0; i < 46; ++i) I[i] = (const float*)d_in[i];
    float* out = (float*)d_out;

    // ---- workspace bump allocator -----------------------------------------
    char* base = (char*)d_ws;
    size_t off = 0;
    auto alloc = [&](size_t bytes) -> void* {
        void* r = base + off;
        off = (off + bytes + 255) & ~(size_t)255;
        return r;
    };
    // persistent small region
    _Float16* staticH = (_Float16*)alloc(BB * SS * 2);
    _Float16* cbankH  = (_Float16*)alloc(NN * SS * 2);
    _Float16* Wpg0H = (_Float16*)alloc(32 * 128 * 2);
    _Float16* Wpg1H = (_Float16*)alloc(128 * 64 * 2);
    _Float16* WvpH  = (_Float16*)alloc(32 * 64 * 2);
    _Float16* Wrg0H = (_Float16*)alloc(128 * 64 * 2);
    _Float16* WlsH  = (_Float16*)alloc(32 * 256 * 2);
    _Float16* Wlc0H = (_Float16*)alloc(768 * 512 * 2);
    _Float16* Wlc1H = (_Float16*)alloc(512 * 256 * 2);
    _Float16* Wk0H  = (_Float16*)alloc(64 * 64 * 2);    // padded 34->64
    _Float16* Wk1H  = (_Float16*)alloc(64 * 64 * 2);
    _Float16* Wpo0H = (_Float16*)alloc(64 * 128 * 2);   // padded 35->64
    _Float16* Wpo1H = (_Float16*)alloc(128 * 128 * 2);
    _Float16* Wpo2H = (_Float16*)alloc(128 * 128 * 2);
    _Float16* Wro0H = (_Float16*)alloc(128 * 128 * 2);
    _Float16* Wro1H = (_Float16*)alloc(128 * 256 * 2);
    _Float16* qh1   = (_Float16*)alloc(BB * 128 * 2);
    _Float16* kh1   = (_Float16*)alloc(NN * 128 * 2);
    _Float16* g_in  = (_Float16*)alloc(BB * 128 * 2);   // [q | pooled]
    _Float16* kmat  = (_Float16*)alloc(NN * 64 * 2);
    _Float16* vmat  = (_Float16*)alloc(NN * 64 * 2);
    float*    scores= (float*)alloc((size_t)BB * NN * 4);
    _Float16* attn  = (_Float16*)alloc((size_t)BB * NN * 2);
    _Float16* h1rho = (_Float16*)alloc(BB * 64 * 2);
    float* alphaB = (float*)alloc(BB * 4);
    float* gB     = (float*)alloc(BB * 4);
    float* Wsum   = (float*)alloc(BB * 4);
    float* vfrac  = (float*)alloc(256);
    _Float16* aggH  = (_Float16*)alloc(BB * 128 * 2);
    _Float16* r1out = (_Float16*)alloc(BB * 128 * 2);
    float* dE    = (float*)alloc(BB * DD * 4);
    float* sproj = (float*)alloc(BB * DD * 4);
    float* w_raw = (float*)alloc((size_t)BN * 4);

    // pair-stage arena (later reused by the local-correction stage)
    const size_t arena0 = off;
    _Float16* Xphi = (_Float16*)alloc((size_t)BN * 64 * 2);
    _Float16* Xkap = (_Float16*)alloc((size_t)BN * 64 * 2);
    _Float16* H1k  = (_Float16*)alloc((size_t)BN * 64 * 2);
    _Float16* H2k  = (_Float16*)alloc((size_t)BN * 64 * 2);
    _Float16* P1   = (_Float16*)alloc((size_t)BN * 128 * 2);
    _Float16* P2   = (_Float16*)alloc((size_t)BN * 128 * 2);
    _Float16* P3   = (_Float16*)alloc((size_t)BN * 128 * 2);
    // loc-stage overlay
    off = arena0;
    _Float16* Xloc   = (_Float16*)alloc((size_t)BT * 768 * 2);
    _Float16* Hloc   = (_Float16*)alloc((size_t)BT * 512 * 2);
    float*    locbuf = (float*)alloc((size_t)BT * 256 * 4);

    auto cvt = [&](const float* src, _Float16* dst, int Ksrc, int N, int Kpad) {
        int tot = Kpad * N;
        k_convert_pad<<<(tot + 255) / 256, 256, 0, stream>>>(src, dst, Ksrc, N, Kpad);
    };

    // ---- weight / input conversion ----------------------------------------
    cvt(I[1],  staticH, 32, SS, 32);            // static (row-major passthrough)
    k_cbank<<<(NN * SS + 255) / 256, 256, 0, stream>>>(I[3], cbankH);
    cvt(I[4],  Wpg0H, 32, 128, 32);
    cvt(I[6],  Wpg1H, 128, 64, 128);
    cvt(I[8],  WvpH,  32, 64, 32);
    cvt(I[10], Wrg0H, 128, 64, 128);
    cvt(I[16], WlsH,  32, 256, 32);
    cvt(I[18], Wlc0H, 768, 512, 768);
    cvt(I[20], Wlc1H, 512, 256, 512);
    cvt(I[24], Wk0H,  34, 64, 64);
    cvt(I[26], Wk1H,  64, 64, 64);
    cvt(I[30], Wpo0H, 35, 128, 64);
    cvt(I[32], Wpo1H, 128, 128, 128);
    cvt(I[34], Wpo2H, 128, 128, 128);
    cvt(I[36], Wro0H, 128, 128, 128);
    cvt(I[38], Wro1H, 128, 256, 128);

    // ---- attention / gating (q,k,v small GEMMs) ---------------------------
    gemm_f16<128,1,false,_Float16><<<dim3(2,1),256,0,stream>>>(staticH,32, Wpg0H,128, I[5],  qh1,128, 32);
    gemm_f16<64,0,false,_Float16> <<<dim3(2,1),256,0,stream>>>(qh1,128,    Wpg1H,64,  I[7],  g_in,128, 128);
    gemm_f16<128,1,false,_Float16><<<dim3(8,1),256,0,stream>>>(cbankH,32,  Wpg0H,128, I[5],  kh1,128, 32);
    gemm_f16<64,0,false,_Float16> <<<dim3(8,1),256,0,stream>>>(kh1,128,    Wpg1H,64,  I[7],  kmat,64, 128);
    gemm_f16<64,0,false,_Float16> <<<dim3(8,1),256,0,stream>>>(cbankH,32,  WvpH,64,   I[9],  vmat,64, 32);
    // scores = q @ k^T (TRB), softmax(/8), pooled = attn @ v -> g_in cols 64..127
    gemm_f16<128,0,true,float>    <<<dim3(2,8),256,0,stream>>>(g_in,128,   kmat,64,   nullptr, scores,NN, 64);
    k_softmax<<<BB,256,0,stream>>>(scores, attn);
    gemm_f16<64,0,false,_Float16> <<<dim3(2,1),256,0,stream>>>(attn,NN,    vmat,64,   nullptr, g_in+64,128, NN);
    gemm_f16<64,1,false,_Float16> <<<dim3(2,1),256,0,stream>>>(g_in,128,   Wrg0H,64,  I[11], h1rho,64, 128);
    k_alpha<<<1,256,0,stream>>>(h1rho, I[12], I[13], alphaB);

    // ---- pairwise kappa / phi_obs -----------------------------------------
    k_tokens<<<BN/256,256,0,stream>>>(I[1], I[3], Xphi, Xkap);
    gemm_f16<64,1,false,_Float16> <<<dim3(2048,1),256,0,stream>>>(Xkap,64, Wk0H,64,  I[25], H1k,64, 64);
    gemm_f16<64,1,false,_Float16> <<<dim3(2048,1),256,0,stream>>>(H1k,64,  Wk1H,64,  I[27], H2k,64, 64);
    k_kappa3<<<BN/256,256,0,stream>>>(H2k, I[28], I[29], I[3], w_raw);
    k_wsum<<<BB,256,0,stream>>>(w_raw, Wsum);
    k_meanm<<<1,256,0,stream>>>(I[3], vfrac);
    k_gate<<<1,256,0,stream>>>(Wsum, vfrac, I[40], I[41], I[42], I[43], I[44], I[45], gB);
    gemm_f16<128,1,false,_Float16><<<dim3(2048,1),256,0,stream>>>(Xphi,64, Wpo0H,128, I[31], P1,128, 64);
    gemm_f16<128,1,false,_Float16><<<dim3(2048,1),256,0,stream>>>(P1,128,  Wpo1H,128, I[33], P2,128, 128);
    gemm_f16<128,0,false,_Float16><<<dim3(2048,1),256,0,stream>>>(P2,128,  Wpo2H,128, I[35], P3,128, 128);
    k_agg<<<BB,128,0,stream>>>(P3, w_raw, Wsum, aggH);
    gemm_f16<128,1,false,_Float16><<<dim3(2,1),256,0,stream>>>(aggH,128,  Wro0H,128, I[37], r1out,128, 128);
    gemm_f16<128,0,false,float>   <<<dim3(2,2),256,0,stream>>>(r1out,128, Wro1H,256, I[39], dE,256, 128);

    // ---- local correction path (dominant GEMMs) ---------------------------
    gemm_f16<128,0,false,float>   <<<dim3(2,2),256,0,stream>>>(staticH,32, WlsH,256, I[17], sproj,256, 32);
    k_xloc<<<BT,256,0,stream>>>(I[0], I[2], I[14], I[15], sproj, Xloc);
    gemm_f16<128,1,false,_Float16><<<dim3(730,4),256,0,stream>>>(Xloc,768, Wlc0H,512, I[19], Hloc,512, 768);
    gemm_f16<128,0,false,float>   <<<dim3(730,2),256,0,stream>>>(Hloc,512, Wlc1H,256, I[21], locbuf,256, 512);
    k_final<<<BT,256,0,stream>>>(I[0], locbuf, dE, alphaB, gB, I[22], I[23], out);
}
